// LSTMModel_4535485464879
// MI455X (gfx1250) — compile-verified, matrix operand order checked by reading, not run on hardware
//
#include <hip/hip_runtime.h>
#include <hip/hip_bf16.h>

// ---------------------------------------------------------------------------
// LSTM (B=256, T=256, D=512, H=1024) + FC(1000) for gfx1250 (MI455X).
// Per-timestep fused GEMM gates = x_t*Wx^T + h*Wh^T + b using
// v_wmma_f32_16x16x32_bf16. A/B tiles staged in LDS (double-buffered):
//   - bf16 tiles (Wx, Wh, h) via Tensor Data Mover (tensor_load_to_lds)
//     with D# LDS-padding (pitch 64 -> 72 elems) to avoid bank conflicts.
//   - x tile staged cooperatively with one fp32->bf16 convert per element.
// Fragments read from LDS with ds_load_b128. Epilogue applies the LSTM cell
// and writes h as bf16 (ping-pong across steps).
// ---------------------------------------------------------------------------

typedef __attribute__((ext_vector_type(16))) __bf16 v16bf;
typedef __attribute__((ext_vector_type(8)))  float  v8f;
typedef __attribute__((ext_vector_type(4)))  unsigned int v4u;
typedef __attribute__((ext_vector_type(8)))  int v8i;
typedef __attribute__((ext_vector_type(4)))  int v4i;

#define HID    1024
#define IND    512
#define NBATCH 256
#define SEQT   256
#define G4H    4096
#define OUTN   1000

#define CHK    64            // K-chunk staged per LDS buffer
#define LDSP   72            // padded LDS row pitch in bf16 elements (144 B)
#define NCH_X  (IND / CHK)   // 8 chunks for the x part
#define NCH_H  (HID / CHK)   // 16 chunks for the h part
#define NCH    (NCH_X + NCH_H)

__device__ __forceinline__ unsigned short f32_to_bf16_bits(float f) {
    union { float f; unsigned u; } v; v.f = f;
    unsigned u = v.u;
    unsigned r = u + 0x7FFFu + ((u >> 16) & 1u);   // round-to-nearest-even
    return (unsigned short)(r >> 16);
}

union Frag16 {
    v16bf v;
    uint4 q[2];
    unsigned short s[16];
};

// Generic pointer -> LDS byte offset (addrspace(3) ptrtoint).
__device__ __forceinline__ unsigned lds_addr_of(const void* p) {
    return (unsigned)(unsigned long long)
        (__attribute__((address_space(3))) const unsigned char*)p;
}

// ---------------------------------------------------------------------------
// TDM: async-load a 64-row x 64-col bf16 tile (row-major, pitch_elems) from
// global to LDS, inserting 16 B of padding every 128 B (LDS pitch = 72).
// D# layout per CDNA5 ISA. Tracked by TENSORcnt.
// ---------------------------------------------------------------------------
__device__ __forceinline__ void tdm_load_tile64(unsigned lds_byte,
                                                const void* gsrc,
                                                unsigned pitch_elems) {
    unsigned long long ga = (unsigned long long)gsrc;
    v4u g0;
    g0[0] = 1u;                                   // count=1 (valid), user mode
    g0[1] = lds_byte;                             // lds_addr
    g0[2] = (unsigned)(ga & 0xFFFFFFFFu);         // global_addr[31:0]
    g0[3] = (unsigned)(ga >> 32) | (2u << 30);    // global_addr[56:32], type=2

    const unsigned dim0 = pitch_elems;            // tensor_dim0 (OOB bound)
    const unsigned dim1 = 8192u;                  // rows bound (never OOB)
    const unsigned long long s0 = pitch_elems;    // tensor_dim0_stride
    v8i g1;
    // data_size=1(2B) | pad_enable | pad_interval=4 (32 DW=128B) | pad_amount=3 (4 DW=16B)
    g1[0] = (int)((1u << 16) | (1u << 20) | (4u << 22) | (3u << 25));
    g1[1] = (int)((dim0 & 0xFFFFu) << 16);                        // tensor_dim0 lo
    g1[2] = (int)(((dim0 >> 16) & 0xFFFFu) | ((dim1 & 0xFFFFu) << 16));
    g1[3] = (int)(((dim1 >> 16) & 0xFFFFu) | (64u << 16));        // tile_dim0=64
    g1[4] = (int)(64u);                                           // tile_dim1=64, tile_dim2=0
    g1[5] = (int)(unsigned)(s0 & 0xFFFFFFFFu);
    g1[6] = (int)(unsigned)((s0 >> 32) & 0xFFFFu);                // stride1=0
    g1[7] = 0;
    v4i z4 = {0, 0, 0, 0};
    v8i z8 = {0, 0, 0, 0, 0, 0, 0, 0};
    __builtin_amdgcn_tensor_load_to_lds(g0, g1, z4, z4, z8, 0);
}

// LDS fragment read (A rows / B cols, ISA 7.12.2 layout), padded pitch LDSP.
__device__ __forceinline__ v16bf lds_frag(const unsigned short* base,
                                          int lane, int k0) {
    const unsigned short* q = base + (lane & 15) * LDSP + k0 + 8 * (lane >> 4);
    Frag16 f;
    f.q[0] = *(const uint4*)q;
    f.q[1] = *(const uint4*)(q + 16);
    return f.v;
}

// Global bf16 fragment read (used by the small FC kernel).
__device__ __forceinline__ v16bf load_frag_bf16(const unsigned short* __restrict__ base,
                                                int row0, int pitch, int k0, int lane) {
    const unsigned short* p = base + (size_t)(row0 + (lane & 15)) * pitch
                            + k0 + 8 * (lane >> 4);
    Frag16 f;
    f.q[0] = *(const uint4*)(p);
    f.q[1] = *(const uint4*)(p + 16);
    return f.v;
}

__device__ __forceinline__ float sigmoidf_fast(float x) {
    return 1.0f / (1.0f + __expf(-x));
}

// ---------------------------------------------------------------------------
// Fused LSTM timestep. Grid (4,16) x 256 threads (8 waves).
// Workgroup: 64 rows x 64 cols(H) x 4 gates. Wave: 32 rows x 16 cols x 4 gates.
// ---------------------------------------------------------------------------
__global__ __launch_bounds__(256) void lstm_step_kernel(
    const float* __restrict__ x,            // [B, T, D] fp32
    const unsigned short* __restrict__ wx,  // [4H, D]  bf16
    const unsigned short* __restrict__ wh,  // [4H, H]  bf16
    const float* __restrict__ bsum,         // [4H] = b_x2h + b_h2h
    const unsigned short* __restrict__ h_in,// [B, H] bf16
    unsigned short* __restrict__ h_out,     // [B, H] bf16
    float* __restrict__ c,                  // [B, H] fp32 (in/out)
    int t)
{
    __shared__ __align__(16) unsigned short Atile[2][64][LDSP];      // 2 x 9 KB
    __shared__ __align__(16) unsigned short Btile[2][4][64][LDSP];   // 2 x 36 KB

    const int tid  = threadIdx.x;
    const int lane = tid & 31;
    const int wave = tid >> 5;
    const int wr = wave >> 2;                       // 0..1
    const int wc = wave & 3;                        // 0..3
    const int m0blk = blockIdx.x * 64;              // batch rows of this WG
    const int n0blk = blockIdx.y * 64;              // H cols of this WG
    const int m0 = m0blk + wr * 32;
    const int n0 = n0blk + wc * 16;

    v8f zero = {0.f,0.f,0.f,0.f,0.f,0.f,0.f,0.f};
    v8f acc[2][4];
#pragma unroll
    for (int mt = 0; mt < 2; ++mt)
#pragma unroll
        for (int g = 0; g < 4; ++g) acc[mt][g] = zero;

    // Cooperative x-tile stage: 64 rows x 64 cols fp32 -> bf16 into Atile.
    // Thread t: row = t/4, 16 cols at (t%4)*16.
    auto stage_x = [&](int ch, int buf) {
        const int r  = tid >> 2;
        const int cg = (tid & 3) * 16;
        const float* src = x + ((size_t)(m0blk + r) * SEQT + (size_t)t) * IND
                         + ch * CHK + cg;
        float4 f0 = ((const float4*)src)[0];
        float4 f1 = ((const float4*)src)[1];
        float4 f2 = ((const float4*)src)[2];
        float4 f3 = ((const float4*)src)[3];
        Frag16 o;
        o.s[0]  = f32_to_bf16_bits(f0.x); o.s[1]  = f32_to_bf16_bits(f0.y);
        o.s[2]  = f32_to_bf16_bits(f0.z); o.s[3]  = f32_to_bf16_bits(f0.w);
        o.s[4]  = f32_to_bf16_bits(f1.x); o.s[5]  = f32_to_bf16_bits(f1.y);
        o.s[6]  = f32_to_bf16_bits(f1.z); o.s[7]  = f32_to_bf16_bits(f1.w);
        o.s[8]  = f32_to_bf16_bits(f2.x); o.s[9]  = f32_to_bf16_bits(f2.y);
        o.s[10] = f32_to_bf16_bits(f2.z); o.s[11] = f32_to_bf16_bits(f2.w);
        o.s[12] = f32_to_bf16_bits(f3.x); o.s[13] = f32_to_bf16_bits(f3.y);
        o.s[14] = f32_to_bf16_bits(f3.z); o.s[15] = f32_to_bf16_bits(f3.w);
        uint4* dst = (uint4*)&Atile[buf][r][cg];
        dst[0] = o.q[0];
        dst[1] = o.q[1];
    };

    // Wave0: issue TDM loads for chunk ch into buffer buf.
    // x phase (ch < NCH_X): 4 B-tiles from Wx. h phase: A from h + 4 B from Wh.
    auto issue_tdm = [&](int ch, int buf) {
        if (ch < NCH_X) {
            const int k = ch * CHK;
#pragma unroll
            for (int g = 0; g < 4; ++g)
                tdm_load_tile64(lds_addr_of(&Btile[buf][g][0][0]),
                                wx + (size_t)(g * HID + n0blk) * IND + k, IND);
        } else {
            const int k = (ch - NCH_X) * CHK;
            tdm_load_tile64(lds_addr_of(&Atile[buf][0][0]),
                            h_in + (size_t)m0blk * HID + k, HID);
#pragma unroll
            for (int g = 0; g < 4; ++g)
                tdm_load_tile64(lds_addr_of(&Btile[buf][g][0][0]),
                                wh + (size_t)(g * HID + n0blk) * HID + k, HID);
        }
    };

    // ---- prefetch chunk 0 ----
    stage_x(0, 0);
    if (wave == 0) issue_tdm(0, 0);

    // ---- main pipeline: 24 chunks, double-buffered ----
    for (int ch = 0; ch < NCH; ++ch) {
        const int cur = ch & 1;
        const int nxt = cur ^ 1;

        if (ch + 1 < NCH) {
            if (ch + 1 < NCH_X) stage_x(ch + 1, nxt);
            if (wave == 0) {
                issue_tdm(ch + 1, nxt);
                if (ch + 1 < NCH_X) __builtin_amdgcn_s_wait_tensorcnt(4);
                else                __builtin_amdgcn_s_wait_tensorcnt(5);
            }
        } else {
            if (wave == 0) __builtin_amdgcn_s_wait_tensorcnt(0);
        }
        __syncthreads();   // cur tiles (coop stores + TDM) visible to all waves

        const unsigned short* At = &Atile[cur][wr * 32][0];
#pragma unroll
        for (int kk = 0; kk < 2; ++kk) {
            const int k0 = kk * 32;
            v16bf a0 = lds_frag(At,             lane, k0);
            v16bf a1 = lds_frag(At + 16 * LDSP, lane, k0);
#pragma unroll
            for (int g = 0; g < 4; ++g) {
                v16bf bm = lds_frag(&Btile[cur][g][wc * 16][0], lane, k0);
                acc[0][g] = __builtin_amdgcn_wmma_f32_16x16x32_bf16(
                    false, a0, false, bm, (short)0, acc[0][g], false, false);
                acc[1][g] = __builtin_amdgcn_wmma_f32_16x16x32_bf16(
                    false, a1, false, bm, (short)0, acc[1][g], false, false);
            }
        }
        __syncthreads();   // done reading cur before it is overwritten
    }

    // ---- Epilogue: LSTM cell. C layout: VGPR e, lane 16*half+nl ->
    // M = e + 8*half, N = nl.
    const int nl   = lane & 15;
    const int half = lane >> 4;
    const int ncol = n0 + nl;
    const float bi = bsum[0 * HID + ncol];
    const float bf = bsum[1 * HID + ncol];
    const float bg = bsum[2 * HID + ncol];
    const float bo = bsum[3 * HID + ncol];

#pragma unroll
    for (int mt = 0; mt < 2; ++mt) {
#pragma unroll
        for (int e = 0; e < 8; ++e) {
            const int m = m0 + mt * 16 + e + 8 * half;
            const size_t idx = (size_t)m * HID + ncol;
            const float iv = sigmoidf_fast(acc[mt][0][e] + bi);
            const float fv = sigmoidf_fast(acc[mt][1][e] + bf);
            const float gv = tanhf(acc[mt][2][e] + bg);
            const float ov = sigmoidf_fast(acc[mt][3][e] + bo);
            const float cn = c[idx] * fv + iv * gv;
            c[idx] = cn;
            h_out[idx] = f32_to_bf16_bits(ov * tanhf(cn));
        }
    }
}

// ---------------------------------------------------------------------------
// Final FC: out[256 x 1000] = h_bf16 @ Wfc^T + b_fc (Wfc zero-padded to 1024
// rows). Grid (4,16) x 256 threads; wave = 32 rows x 16 cols, K = 1024.
// ---------------------------------------------------------------------------
__global__ __launch_bounds__(256) void fc_kernel(
    const unsigned short* __restrict__ h,     // [B, H] bf16
    const unsigned short* __restrict__ wfc,   // [1024, H] bf16 (padded)
    const float* __restrict__ bfc,            // [1000]
    float* __restrict__ out)                  // [B, 1000]
{
    const int lane = threadIdx.x & 31;
    const int wave = threadIdx.x >> 5;
    const int wr = wave >> 2;
    const int wc = wave & 3;
    const int m0 = blockIdx.x * 64 + wr * 32;
    const int n0 = blockIdx.y * 64 + wc * 16;

    v8f zero = {0.f,0.f,0.f,0.f,0.f,0.f,0.f,0.f};
    v8f acc0 = zero, acc1 = zero;

    for (int k = 0; k < HID; k += 32) {
        v16bf a0 = load_frag_bf16(h, m0,      HID, k, lane);
        v16bf a1 = load_frag_bf16(h, m0 + 16, HID, k, lane);
        v16bf bm = load_frag_bf16(wfc, n0, HID, k, lane);
        acc0 = __builtin_amdgcn_wmma_f32_16x16x32_bf16(
            false, a0, false, bm, (short)0, acc0, false, false);
        acc1 = __builtin_amdgcn_wmma_f32_16x16x32_bf16(
            false, a1, false, bm, (short)0, acc1, false, false);
    }

    const int nl   = lane & 15;
    const int half = lane >> 4;
    const int ncol = n0 + nl;
    if (ncol < OUTN) {
        const float bv = bfc[ncol];
#pragma unroll
        for (int e = 0; e < 8; ++e) {
            const int ma = m0 + e + 8 * half;
            const int mb = ma + 16;
            out[(size_t)ma * OUTN + ncol] = acc0[e] + bv;
            out[(size_t)mb * OUTN + ncol] = acc1[e] + bv;
        }
    }
}

// ---------------------------------------------------------------------------
// Setup kernels
// ---------------------------------------------------------------------------
__global__ void cvt_bf16_kernel(const float* __restrict__ src,
                                unsigned short* __restrict__ dst, int n) {
    int i = blockIdx.x * blockDim.x + threadIdx.x;
    if (i < n) dst[i] = f32_to_bf16_bits(src[i]);
}

__global__ void cvt_pad_wfc_kernel(const float* __restrict__ src,
                                   unsigned short* __restrict__ dst) {
    int i = blockIdx.x * blockDim.x + threadIdx.x;   // over 1024*1024
    int r = i >> 10;
    dst[i] = (r < OUTN) ? f32_to_bf16_bits(src[i]) : (unsigned short)0;
}

__global__ void bias_sum_kernel(const float* __restrict__ a,
                                const float* __restrict__ b,
                                float* __restrict__ o, int n) {
    int i = blockIdx.x * blockDim.x + threadIdx.x;
    if (i < n) o[i] = a[i] + b[i];
}

__global__ void zero_f32_kernel(float* __restrict__ p, int n) {
    int i = blockIdx.x * blockDim.x + threadIdx.x;
    if (i < n) p[i] = 0.0f;
}

__global__ void zero_u16_kernel(unsigned short* __restrict__ p, int n) {
    int i = blockIdx.x * blockDim.x + threadIdx.x;
    if (i < n) p[i] = 0;
}

// ---------------------------------------------------------------------------
extern "C" void kernel_launch(void* const* d_in, const int* in_sizes, int n_in,
                              void* d_out, int out_size, void* d_ws, size_t ws_size,
                              hipStream_t stream) {
    (void)in_sizes; (void)n_in; (void)out_size; (void)ws_size;

    const float* x   = (const float*)d_in[0];   // [B,T,D]
    const float* wx  = (const float*)d_in[1];   // [4H,D]
    const float* bx  = (const float*)d_in[2];   // [4H]
    const float* wh  = (const float*)d_in[3];   // [4H,H]
    const float* bh  = (const float*)d_in[4];   // [4H]
    const float* wfc = (const float*)d_in[5];   // [1000,H]
    const float* bfc = (const float*)d_in[6];   // [1000]
    float* out = (float*)d_out;

    // Workspace carve-up (~17 MB, all 256B aligned)
    char* ws = (char*)d_ws;
    unsigned short* wx_bf  = (unsigned short*)ws; ws += (size_t)G4H * IND * 2;   // 4 MB
    unsigned short* wh_bf  = (unsigned short*)ws; ws += (size_t)G4H * HID * 2;   // 8 MB
    unsigned short* wfc_bf = (unsigned short*)ws; ws += (size_t)1024 * HID * 2;  // 2 MB
    float*          bsum   = (float*)ws;          ws += (size_t)G4H * 4;         // 16 KB
    float*          c      = (float*)ws;          ws += (size_t)NBATCH * HID * 4;// 1 MB
    unsigned short* hbf0   = (unsigned short*)ws; ws += (size_t)NBATCH * HID * 2;// 512 KB
    unsigned short* hbf1   = (unsigned short*)ws;                                 // 512 KB

    // One-time conversions / init
    cvt_bf16_kernel<<<(G4H * IND + 255) / 256, 256, 0, stream>>>(wx, wx_bf, G4H * IND);
    cvt_bf16_kernel<<<(G4H * HID + 255) / 256, 256, 0, stream>>>(wh, wh_bf, G4H * HID);
    cvt_pad_wfc_kernel<<<(1024 * HID) / 256, 256, 0, stream>>>(wfc, wfc_bf);
    bias_sum_kernel<<<(G4H + 255) / 256, 256, 0, stream>>>(bx, bh, bsum, G4H);
    zero_f32_kernel<<<(NBATCH * HID + 255) / 256, 256, 0, stream>>>(c, NBATCH * HID);
    zero_u16_kernel<<<(NBATCH * HID + 255) / 256, 256, 0, stream>>>(hbf0, NBATCH * HID);

    // Recurrent scan: 256 fused GEMM+cell steps (h ping-pong)
    dim3 grid(NBATCH / 64, HID / 64);   // (4,16)
    for (int t = 0; t < SEQT; ++t) {
        const unsigned short* hin = (t & 1) ? hbf1 : hbf0;
        unsigned short*      hout = (t & 1) ? hbf0 : hbf1;
        lstm_step_kernel<<<grid, 256, 0, stream>>>(x, wx_bf, wh_bf, bsum,
                                                   hin, hout, c, t);
    }

    // Final FC reads hbf0 (t=255 writes hbf0)
    fc_kernel<<<grid, 256, 0, stream>>>(hbf0, wfc_bf, bfc, out);
}
